// LayoutEarlyJoinGConv_764504179149
// MI455X (gfx1250) — matrix operand order repaired and checked by reading, not compiled, not run on hardware
//
#include <hip/hip_runtime.h>

typedef __attribute__((ext_vector_type(16))) _Float16 v16h;
typedef __attribute__((ext_vector_type(8)))  float    v8f;

#define DEV __device__ __forceinline__

static constexpr int NN = 200000;   // nodes
static constexpr int EE = 800000;   // edges
static constexpr int NB = 16;       // graphs

// ---------------------------------------------------------------------------
// Async global->LDS staging (gfx1250 GLOBAL_LOAD_ASYNC_TO_LDS_B128, ASYNCcnt).
// Builtin signature (from hipcc diagnostics): arg0 = int32x4* in AS1 (global),
// arg1 = LDS dest, arg2/arg3 = imm offset / cpol.
// Guarded: falls back to a plain cooperative copy if the builtin is absent.
// ---------------------------------------------------------------------------
#if defined(__gfx1250__) && __has_builtin(__builtin_amdgcn_global_load_async_to_lds_b128)
#define USE_ASYNC_LDS 1
typedef int v4i_vs __attribute__((vector_size(16)));
typedef __attribute__((address_space(1))) v4i_vs GV4;
typedef __attribute__((address_space(3))) v4i_vs LV4;
#else
#define USE_ASYNC_LDS 0
#endif

// Issue the copy of `bytes` (multiple of 4096) from W to sW. 256 threads.
DEV void stage_weights_issue(const _Float16* W, _Float16* sW, int bytes,
                             int tid) {
  const int n16 = bytes >> 4;
#if USE_ASYNC_LDS
  GV4* g = (GV4*)(uintptr_t)W;
  LV4* l = (LV4*)(uint32_t)(uintptr_t)sW;   // low 32 bits = LDS offset
  for (int i = tid; i < n16; i += 256) {
    __builtin_amdgcn_global_load_async_to_lds_b128(g + i, l + i, 0, 0);
  }
#else
  for (int i = tid; i < n16; i += 256)
    ((uint4*)sW)[i] = ((const uint4*)W)[i];
#endif
}

DEV void stage_weights_wait() {
#if USE_ASYNC_LDS
  asm volatile("s_wait_asynccnt 0" ::: "memory");
#endif
}

// ---------------------------------------------------------------------------
// WMMA fragment helpers (CDNA5 V_WMMA_F32_16X16X32_F16, wave32, ISA §7.12.2)
// A: 16x32 f16, lane m = lane&15, k = (lane/16)*16 + i (contiguous order chosen here, not the ISA table order)
// B: 32x16 f16, lane n = lane&15, k = (lane/16)*16 + i (same order as A, which makes it valid)
// C/D: 16x16 f32, row m = vgpr + (lane/16)*8, col n = lane&15
// ---------------------------------------------------------------------------
DEV v16h load_a_frag(const _Float16* A, int lda, int lane) {
  const int m    = lane & 15;
  const int koff = (lane >> 4) << 4;
  v16h a;
#pragma unroll
  for (int i = 0; i < 16; ++i) {
    const int k = koff + i;
    a[i] = A[m * lda + k];
  }
  return a;
}

DEV v16h load_b_frag(const _Float16* B, int ldb, int lane) {
  const int n    = lane & 15;
  const int koff = (lane >> 4) << 4;
  v16h b;
#pragma unroll
  for (int i = 0; i < 16; ++i) {
    b[i] = B[(koff + i) * ldb + n];
  }
  return b;
}

DEV v8f wmma_f16(v16h a, v16h b, v8f c) {
  return __builtin_amdgcn_wmma_f32_16x16x32_f16(false, a, false, b, (short)0, c,
                                                false, false);
}

DEV void atomic_max_f32(float* addr, float val) {
  unsigned int* ua  = (unsigned int*)addr;
  unsigned int  old = *ua;
  while (__uint_as_float(old) < val) {
    unsigned int assumed = old;
    old = atomicCAS(ua, assumed, __float_as_uint(val));
    if (old == assumed) break;
  }
}

// ---------------------------------------------------------------------------
// Utility kernels
// ---------------------------------------------------------------------------
__global__ void fill_f32_kernel(float* __restrict__ p, float v, long long n) {
  long long i = (long long)blockIdx.x * blockDim.x + threadIdx.x;
  if (i < n) p[i] = v;
}

// fp32 -> fp16 row-major weight convert with zero row padding
__global__ void cvt_f16_kernel(const float* __restrict__ src,
                               _Float16* __restrict__ dst,
                               int srows, int cols, int drows) {
  int i = blockIdx.x * blockDim.x + threadIdx.x;
  if (i >= drows * cols) return;
  int r  = i / cols;
  dst[i] = (r < srows) ? (_Float16)src[i] : (_Float16)0.0f;
}

// in-degree accumulate + invert
__global__ void deg_scatter_kernel(const int* __restrict__ ei,
                                   float* __restrict__ deg) {
  int e = blockIdx.x * blockDim.x + threadIdx.x;
  if (e < EE) atomicAdd(&deg[ei[EE + e]], 1.0f);
}

__global__ void deg_finalize_kernel(float* __restrict__ deg) {
  int i = blockIdx.x * blockDim.x + threadIdx.x;
  if (i < NN) deg[i] = 1.0f / fmaxf(deg[i], 1.0f);
}

// ---------------------------------------------------------------------------
// Fused embed-gather + concat + input linear (K padded 193->224) + bias + relu
//   X[n, 0:128] = relu(concat(feat, type_emb, op_emb, cfg) @ W0 + b0)
// 64 nodes / workgroup, 8 waves, each wave owns 4 of the 4x8 16x16 out tiles.
// W0 (224x128 f16 = 57KB) async-staged into LDS, overlapped with gather.
// ---------------------------------------------------------------------------
__global__ __launch_bounds__(256) void input_mlp_kernel(
    const float* __restrict__ node_feat,   // N x 140 (last col = type id)
    const float* __restrict__ cfg,         // N x 18
    const int* __restrict__ opcode,        // N
    const float* __restrict__ op_emb,      // 120 x 32
    const float* __restrict__ type_emb,    // 8 x 4
    const _Float16* __restrict__ W0,       // 224 x 128 (fp16, rows >=193 zero)
    const float* __restrict__ b0,          // 128
    float* __restrict__ X)                 // N x 128
{
  __shared__ _Float16 sA[64][232];         // 64 x 224 tile, padded stride
  __shared__ _Float16 sW[224 * 128];       // weight panel
  const int row0 = blockIdx.x * 64;
  const int tid  = threadIdx.x;

  stage_weights_issue(W0, sW, 224 * 128 * 2, tid);   // async DMA in flight

  for (int idx = tid; idx < 64 * 224; idx += 256) {  // overlapped VALU work
    const int r = idx / 224, c = idx - r * 224;
    const int node = row0 + r;
    float v;
    if (c < 139) {
      v = node_feat[node * 140 + c];
    } else if (c < 143) {
      const int ty = (int)node_feat[node * 140 + 139];
      v = type_emb[ty * 4 + (c - 139)];
    } else if (c < 175) {
      const int op = opcode[node];
      v = op_emb[op * 32 + (c - 143)];
    } else if (c < 193) {
      v = cfg[node * 18 + (c - 175)];
    } else {
      v = 0.0f;                            // K padding
    }
    sA[r][c] = (_Float16)v;
  }
  stage_weights_wait();
  __syncthreads();

  const int lane = tid & 31, wave = tid >> 5;
  for (int t = wave; t < 32; t += 8) {     // 4 mtiles x 8 ntiles
    const int mt = t >> 3, nt = t & 7;
    v8f acc = {};
#pragma unroll
    for (int ks = 0; ks < 7; ++ks) {       // K = 224 = 7 * 32
      v16h a = load_a_frag(&sA[mt * 16][ks * 32], 232, lane);
      v16h b = load_b_frag(&sW[(ks * 32) * 128 + nt * 16], 128, lane);
      acc = wmma_f16(a, b, acc);
    }
    const int m = (lane >> 4) << 3;
    const int n = nt * 16 + (lane & 15);
    const float bv = b0[n];
#pragma unroll
    for (int r = 0; r < 8; ++r) {
      float v = acc[r] + bv;
      X[(row0 + mt * 16 + m + r) * 128 + n] = v > 0.0f ? v : 0.0f;
    }
  }
}

// ---------------------------------------------------------------------------
// xp = relu(X @ Wp + bp)  (SAGE project).  X stride fixed at 128.
// ---------------------------------------------------------------------------
template <int KD, int NOUT>
__global__ __launch_bounds__(256) void project_kernel(
    const float* __restrict__ X, const _Float16* __restrict__ W,
    const float* __restrict__ bias, float* __restrict__ Y)
{
  __shared__ _Float16 sA[64][KD + 8];
  __shared__ _Float16 sW[KD * NOUT];
  const int row0 = blockIdx.x * 64;
  const int tid  = threadIdx.x;

  stage_weights_issue(W, sW, KD * NOUT * 2, tid);

  for (int idx = tid; idx < 64 * KD; idx += 256) {
    const int r = idx / KD, c = idx - r * KD;
    sA[r][c] = (_Float16)X[(row0 + r) * 128 + c];
  }
  stage_weights_wait();
  __syncthreads();

  const int lane = tid & 31, wave = tid >> 5;
  constexpr int NT = NOUT / 16;
  for (int t = wave; t < 4 * NT; t += 8) {
    const int mt = t / NT, nt = t % NT;
    v8f acc = {};
#pragma unroll
    for (int ks = 0; ks < KD / 32; ++ks) {
      v16h a = load_a_frag(&sA[mt * 16][ks * 32], KD + 8, lane);
      v16h b = load_b_frag(&sW[(ks * 32) * NOUT + nt * 16], NOUT, lane);
      acc = wmma_f16(a, b, acc);
    }
    const int m = (lane >> 4) << 3;
    const int n = nt * 16 + (lane & 15);
    const float bv = bias[n];
#pragma unroll
    for (int r = 0; r < 8; ++r) {
      float v = acc[r] + bv;
      Y[(row0 + mt * 16 + m + r) * 128 + n] = v > 0.0f ? v : 0.0f;
    }
  }
}

// ---------------------------------------------------------------------------
// Mean-aggregate edge scatter: agg[dst] += xp[src]  (float4 per thread)
// ---------------------------------------------------------------------------
template <int KD>
__global__ void edge_scatter_kernel(const int* __restrict__ ei,
                                    const float* __restrict__ xp,
                                    float* __restrict__ agg) {
  constexpr int CH = KD / 4;
  const int idx = blockIdx.x * blockDim.x + threadIdx.x;
  if (idx >= EE * CH) return;
  const int e = idx / CH;
  const int c = (idx - e * CH) * 4;
  const int s = ei[e], d = ei[EE + e];
  const float4 v = *(const float4*)&xp[s * 128 + c];
  atomicAdd(&agg[d * 128 + c + 0], v.x);
  atomicAdd(&agg[d * 128 + c + 1], v.y);
  atomicAdd(&agg[d * 128 + c + 2], v.z);
  atomicAdd(&agg[d * 128 + c + 3], v.w);
}

// ---------------------------------------------------------------------------
// out = L2norm( [agg*deg_inv | x] @ [wl ; wr] + bl ).  Concat-K WMMA GEMM,
// row-normalized through LDS, written in place to X (cols 0..63).
// ---------------------------------------------------------------------------
template <int KD>
__global__ __launch_bounds__(256) void combine_kernel(
    const float* __restrict__ Agg, const float* __restrict__ Xin,
    const float* __restrict__ deg_inv, const _Float16* __restrict__ Wc,
    const float* __restrict__ bl, float* __restrict__ Y)
{
  __shared__ _Float16 sA[64][2 * KD + 8];
  __shared__ _Float16 sW[2 * KD * 64];
  __shared__ float    sOut[64][65];
  const int row0 = blockIdx.x * 64;
  const int tid  = threadIdx.x;

  stage_weights_issue(Wc, sW, 2 * KD * 64 * 2, tid);

  for (int idx = tid; idx < 64 * 2 * KD; idx += 256) {
    const int r = idx / (2 * KD), c = idx - r * (2 * KD);
    const int node = row0 + r;
    float v;
    if (c < KD) v = Agg[node * 128 + c] * deg_inv[node];
    else        v = Xin[node * 128 + (c - KD)];
    sA[r][c] = (_Float16)v;
  }
  stage_weights_wait();
  __syncthreads();

  const int lane = tid & 31, wave = tid >> 5;
  for (int t = wave; t < 16; t += 8) {     // 4 mtiles x 4 ntiles (NOUT=64)
    const int mt = t >> 2, nt = t & 3;
    v8f acc = {};
#pragma unroll
    for (int ks = 0; ks < (2 * KD) / 32; ++ks) {
      v16h a = load_a_frag(&sA[mt * 16][ks * 32], 2 * KD + 8, lane);
      v16h b = load_b_frag(&sW[(ks * 32) * 64 + nt * 16], 64, lane);
      acc = wmma_f16(a, b, acc);
    }
    const int m = (lane >> 4) << 3;
    const int n = nt * 16 + (lane & 15);
    const float bv = bl[n];
#pragma unroll
    for (int r = 0; r < 8; ++r) sOut[mt * 16 + m + r][n] = acc[r] + bv;
  }
  __syncthreads();

  if (tid < 64) {                          // per-row L2 normalize
    float s = 0.0f;
#pragma unroll 8
    for (int c = 0; c < 64; ++c) { float v = sOut[tid][c]; s += v * v; }
    const float inv = 1.0f / fmaxf(sqrtf(s), 1e-12f);
#pragma unroll 8
    for (int c = 0; c < 64; ++c)
      Y[(row0 + tid) * 128 + c] = sOut[tid][c] * inv;
  }
}

// ---------------------------------------------------------------------------
// Per-graph pooling (max + sum + count) and finalize head
// ---------------------------------------------------------------------------
__global__ void pool_scatter_kernel(const float* __restrict__ X,
                                    const int* __restrict__ batch,
                                    float* __restrict__ pmax,
                                    float* __restrict__ psum,
                                    float* __restrict__ cnt) {
  const int idx = blockIdx.x * blockDim.x + threadIdx.x;
  if (idx >= NN * 64) return;
  const int node = idx >> 6, c = idx & 63;
  const int g = batch[node];
  const float v = X[node * 128 + c];
  atomicAdd(&psum[g * 64 + c], v);
  atomic_max_f32(&pmax[g * 64 + c], v);
  if (c == 0) atomicAdd(&cnt[g], 1.0f);
}

__global__ __launch_bounds__(64) void pool_finalize_kernel(
    const float* __restrict__ pmax, const float* __restrict__ psum,
    const float* __restrict__ cnt, const float* __restrict__ pw,
    const float* __restrict__ pb, float* __restrict__ out) {
  __shared__ float sred[64];
  const int b = blockIdx.x, c = threadIdx.x;
  const float g = pmax[b * 64 + c] + psum[b * 64 + c] / fmaxf(cnt[b], 1.0f);
  sred[c] = g * g;
  __syncthreads();
  for (int s = 32; s > 0; s >>= 1) {
    if (c < s) sred[c] += sred[c + s];
    __syncthreads();
  }
  const float inv = 1.0f / fmaxf(sqrtf(sred[0]), 1e-12f);
  __syncthreads();
  sred[c] = g * inv * pw[c];
  __syncthreads();
  for (int s = 32; s > 0; s >>= 1) {
    if (c < s) sred[c] += sred[c + s];
    __syncthreads();
  }
  if (c == 0) out[b] = sred[0] + pb[0];
}

// ---------------------------------------------------------------------------
// Launch orchestration
// ---------------------------------------------------------------------------
extern "C" void kernel_launch(void* const* d_in, const int* in_sizes, int n_in,
                              void* d_out, int out_size, void* d_ws,
                              size_t ws_size, hipStream_t stream) {
  const float* node_feat = (const float*)d_in[0];
  const float* cfg       = (const float*)d_in[1];
  const int*   opcode    = (const int*)d_in[2];
  const int*   ei        = (const int*)d_in[3];
  const int*   batch     = (const int*)d_in[4];
  const float* op_emb    = (const float*)d_in[5];
  const float* type_emb  = (const float*)d_in[6];
  const float* lin_w     = (const float*)d_in[7];
  const float* lin_b     = (const float*)d_in[8];
  const float* post_w    = (const float*)d_in[9];
  const float* post_b    = (const float*)d_in[10];
  const float* wp[3] = {(const float*)d_in[11], (const float*)d_in[16], (const float*)d_in[21]};
  const float* bp[3] = {(const float*)d_in[12], (const float*)d_in[17], (const float*)d_in[22]};
  const float* wl[3] = {(const float*)d_in[13], (const float*)d_in[18], (const float*)d_in[23]};
  const float* blv[3] = {(const float*)d_in[14], (const float*)d_in[19], (const float*)d_in[24]};
  const float* wr[3] = {(const float*)d_in[15], (const float*)d_in[20], (const float*)d_in[25]};

  // ---- workspace layout (bytes) ----
  char* ws = (char*)d_ws;
  _Float16* W0  = (_Float16*)(ws + 0);            // 224x128
  _Float16* WP0 = (_Float16*)(ws + 57344);        // 128x128
  _Float16* WC0 = (_Float16*)(ws + 90112);        // 256x64  [wl0 ; wr0]
  _Float16* WP1 = (_Float16*)(ws + 122880);       // 64x64
  _Float16* WC1 = (_Float16*)(ws + 131072);       // 128x64
  _Float16* WP2 = (_Float16*)(ws + 147456);       // 64x64
  _Float16* WC2 = (_Float16*)(ws + 155648);       // 128x64
  float* deg  = (float*)(ws + 172032);            // N
  float* pmax = (float*)(ws + 972032);            // B x 64
  float* psum = (float*)(ws + 976128);            // B x 64
  float* cnt  = (float*)(ws + 980224);            // B
  float* buf0 = (float*)(ws + 1048576);           // N x 128  (x)
  float* buf1 = (float*)(ws + 103448576ll);       // N x 128  (xp)
  float* buf2 = (float*)(ws + 205848576ll);       // N x 128  (agg)
  float* outp = (float*)d_out;                    // B x 1

  const int NWG = NN / 64;                        // 3125, exact

  auto cvt = [&](const float* s, _Float16* d, int sr, int cl, int dr) {
    cvt_f16_kernel<<<(dr * cl + 255) / 256, 256, 0, stream>>>(s, d, sr, cl, dr);
  };

  // ---- weight conversion to fp16 arena ----
  cvt(lin_w, W0, 193, 128, 224);
  cvt(wp[0], WP0, 128, 128, 128);
  cvt(wl[0], WC0, 128, 64, 128);
  cvt(wr[0], WC0 + 128 * 64, 128, 64, 128);
  cvt(wp[1], WP1, 64, 64, 64);
  cvt(wl[1], WC1, 64, 64, 64);
  cvt(wr[1], WC1 + 64 * 64, 64, 64, 64);
  cvt(wp[2], WP2, 64, 64, 64);
  cvt(wl[2], WC2, 64, 64, 64);
  cvt(wr[2], WC2 + 64 * 64, 64, 64, 64);

  // ---- degree inverse ----
  fill_f32_kernel<<<(NN + 255) / 256, 256, 0, stream>>>(deg, 0.0f, NN);
  deg_scatter_kernel<<<(EE + 255) / 256, 256, 0, stream>>>(ei, deg);
  deg_finalize_kernel<<<(NN + 255) / 256, 256, 0, stream>>>(deg);

  // ---- fused gather + input MLP ----
  input_mlp_kernel<<<NWG, 256, 0, stream>>>(node_feat, cfg, opcode, op_emb,
                                            type_emb, W0, lin_b, buf0);

  // ---- SAGE layer 0 (in=128) ----
  project_kernel<128, 128><<<NWG, 256, 0, stream>>>(buf0, WP0, bp[0], buf1);
  fill_f32_kernel<<<((long long)NN * 128 + 255) / 256, 256, 0, stream>>>(
      buf2, 0.0f, (long long)NN * 128);
  edge_scatter_kernel<128><<<(EE * 32 + 255) / 256, 256, 0, stream>>>(ei, buf1, buf2);
  combine_kernel<128><<<NWG, 256, 0, stream>>>(buf2, buf0, deg, WC0, blv[0], buf0);

  // ---- SAGE layers 1,2 (in=64) ----
  project_kernel<64, 64><<<NWG, 256, 0, stream>>>(buf0, WP1, bp[1], buf1);
  fill_f32_kernel<<<((long long)NN * 128 + 255) / 256, 256, 0, stream>>>(
      buf2, 0.0f, (long long)NN * 128);
  edge_scatter_kernel<64><<<(EE * 16 + 255) / 256, 256, 0, stream>>>(ei, buf1, buf2);
  combine_kernel<64><<<NWG, 256, 0, stream>>>(buf2, buf0, deg, WC1, blv[1], buf0);

  project_kernel<64, 64><<<NWG, 256, 0, stream>>>(buf0, WP2, bp[2], buf1);
  fill_f32_kernel<<<((long long)NN * 128 + 255) / 256, 256, 0, stream>>>(
      buf2, 0.0f, (long long)NN * 128);
  edge_scatter_kernel<64><<<(EE * 16 + 255) / 256, 256, 0, stream>>>(ei, buf1, buf2);
  combine_kernel<64><<<NWG, 256, 0, stream>>>(buf2, buf0, deg, WC2, blv[2], buf0);

  // ---- pooling + head ----
  fill_f32_kernel<<<(NB * 64 + 255) / 256, 256, 0, stream>>>(pmax, -3.4e38f, NB * 64);
  fill_f32_kernel<<<(NB * 64 + 255) / 256, 256, 0, stream>>>(psum, 0.0f, NB * 64);
  fill_f32_kernel<<<1, 256, 0, stream>>>(cnt, 0.0f, NB);
  pool_scatter_kernel<<<((long long)NN * 64 + 255) / 256, 256, 0, stream>>>(
      buf0, batch, pmax, psum, cnt);
  pool_finalize_kernel<<<NB, 64, 0, stream>>>(pmax, psum, cnt, post_w, post_b, outp);
}
